// MultiAttention_86053964743323
// MI455X (gfx1250) — matrix operand order repaired and checked by reading, not compiled, not run on hardware
//
#include <hip/hip_runtime.h>

// Problem sizes (compile-time constants from the reference)
#define B_  2
#define S_  2048
#define IN_ 512
#define HD_ 32
#define H_  16

typedef __attribute__((ext_vector_type(16))) __bf16 v16bf;
typedef __attribute__((ext_vector_type(8)))  float  v8f;
typedef unsigned int v4u __attribute__((ext_vector_type(4), aligned(16), may_alias));

union FragU { v16bf f; v4u q[2]; };

__device__ __forceinline__ unsigned short f2bf(float x) {
  unsigned int u = __float_as_uint(x);
  u += 0x7FFFu + ((u >> 16) & 1u);        // round-to-nearest-even
  return (unsigned short)(u >> 16);
}

__device__ __forceinline__ v8f wmma_bf16(v16bf a, v16bf b, v8f c) {
  // D = A(16x32 bf16) x B(32x16 bf16) + C(16x16 f32)
  return __builtin_amdgcn_wmma_f32_16x16x32_bf16(false, a, false, b, (short)0, c, false, false);
}

// A fragment: rows (row0 + lane&15) of a row-major [*, ld] bf16 matrix,
// K-tile of 32 starting at col0.  Lane group g=lane>>4 holds K offsets
// {g*16..g*16+15} (contiguous 32B, same K order as load_fragB).
__device__ __forceinline__ v16bf load_fragA(const unsigned short* base, int row0,
                                            int ldElems, int col0, int lane) {
  const int r = lane & 15, g = lane >> 4;
  const char* p = (const char*)(base + (size_t)(row0 + r) * ldElems + col0);
  FragU u;
  u.q[0] = *(const v4u*)(p + g * 32);
  u.q[1] = *(const v4u*)(p + g * 32 + 16);
  return u.f;
}

// B fragment built from B^T stored row-major [N, K]: lane holds column
// N=lane&15 of B; lane group g holds K = g*16 .. g*16+15 (contiguous 32B).
__device__ __forceinline__ v16bf load_fragB(const unsigned short* baseT, int n0,
                                            int ldElems, int k0, int lane) {
  const int nn = lane & 15, g = lane >> 4;
  const char* p = (const char*)(baseT + (size_t)(n0 + nn) * ldElems + k0) + g * 32;
  FragU u;
  u.q[0] = *(const v4u*)(p);
  u.q[1] = *(const v4u*)(p + 16);
  return u.f;
}

// ---------------- Stage 0: precision conversion / layout ----------------

__global__ void cvt_k(const float* __restrict__ src, unsigned short* __restrict__ dst, int n) {
  int i = blockIdx.x * 256 + threadIdx.x;
  if (i < n) dst[i] = f2bf(src[i]);
}

// src [H][IN][HD] fp32  ->  dst [H][HD][IN] bf16  (B^T layout for WMMA)
__global__ void trq_k(const float* __restrict__ src, unsigned short* __restrict__ dst, int n) {
  int i = blockIdx.x * 256 + threadIdx.x;
  if (i >= n) return;
  int in   = i & (IN_ - 1);
  int rest = i >> 9;              // IN_=512
  int d    = rest & (HD_ - 1);
  int h    = rest >> 5;           // HD_=32
  dst[i] = f2bf(src[((size_t)h * IN_ + in) * HD_ + d]);
}

// ---------------- Stage 1: fused per-head linear + QKV projection --------
// Grid: B*H*(S/64) blocks of 512 threads (16 waves); 64-row s-tile for 4x
// better Wl reuse out of L2 than a 16-row tile.
// K loops kept rolled (#pragma unroll 1): one iteration already exposes
// 8 wmma + 12 clause-able b128 loads; full unroll caused VGPR spills.
__global__ __launch_bounds__(512) void qkv_k(
    const unsigned short* __restrict__ xb,  const unsigned short* __restrict__ wlb,
    const float* __restrict__ bl,
    const unsigned short* __restrict__ wqt, const unsigned short* __restrict__ wkt,
    const unsigned short* __restrict__ wvt,
    unsigned short* __restrict__ Qb, unsigned short* __restrict__ Kb,
    unsigned short* __restrict__ Vtb) {
  __shared__ __align__(16) unsigned short xh[64 * IN_];   // 64 KB
  const int lane = threadIdx.x & 31, wave = threadIdx.x >> 5;
  const int g = lane >> 4, n = lane & 15;
  const int stile = blockIdx.x & 31;        // S_/64
  const int bh = blockIdx.x >> 5;
  const int h = bh & (H_ - 1);
  const int s0 = stile * 64;
  const unsigned short* xrow = xb + (size_t)((bh >> 4) * S_ + s0) * IN_;
  const unsigned short* wl   = wlb + (size_t)h * IN_ * IN_;
  const int mp = wave >> 3;                 // 0..1
  const int og = wave & 7;                  // 0..7

  v8f acc[2][4] = {};
#pragma unroll 1
  for (int kt = 0; kt < 16; ++kt) {
    v16bf a0 = load_fragA(xrow, 32 * mp,      IN_, kt * 32, lane);
    v16bf a1 = load_fragA(xrow, 32 * mp + 16, IN_, kt * 32, lane);
#pragma unroll
    for (int j = 0; j < 4; ++j) {
      v16bf bb = load_fragB(wl, og * 64 + j * 16, IN_, kt * 32, lane);
      acc[0][j] = wmma_bf16(a0, bb, acc[0][j]);
      acc[1][j] = wmma_bf16(a1, bb, acc[1][j]);
    }
  }
#pragma unroll
  for (int j = 0; j < 4; ++j) {
    const int o = og * 64 + j * 16 + n;
    const float bias = bl[h * IN_ + o];
#pragma unroll
    for (int mt = 0; mt < 2; ++mt)
#pragma unroll
      for (int v = 0; v < 8; ++v)
        xh[(32 * mp + 16 * mt + v + 8 * g) * IN_ + o] = f2bf(acc[mt][j][v] + bias);
  }
  __syncthreads();

  if (wave < 12) {                // 3 matrices x 2 d-tiles x 2 row-pairs
    const int which = wave >> 2;
    const int dbase = ((wave >> 1) & 1) * 16;
    const int mp2   = wave & 1;
    const unsigned short* wt =
        (which == 0 ? wqt : which == 1 ? wkt : wvt) + ((size_t)h * HD_ + dbase) * IN_;
    v8f a2[2] = {};
#pragma unroll 1
    for (int kt = 0; kt < 16; ++kt) {
      v16bf bb = load_fragB(wt, 0, IN_, kt * 32, lane);
      v16bf aA = load_fragA(xh, 32 * mp2,      IN_, kt * 32, lane);
      v16bf aB = load_fragA(xh, 32 * mp2 + 16, IN_, kt * 32, lane);
      a2[0] = wmma_bf16(aA, bb, a2[0]);
      a2[1] = wmma_bf16(aB, bb, a2[1]);
    }
    if (which < 2) {
      unsigned short* dst = (which == 0 ? Qb : Kb)
          + ((size_t)bh * S_ + s0 + 32 * mp2) * HD_ + dbase + n;
#pragma unroll
      for (int mt = 0; mt < 2; ++mt)
#pragma unroll
        for (int v = 0; v < 8; ++v)
          dst[(16 * mt + v + 8 * g) * HD_] = f2bf(a2[mt][v]);
    } else {                                 // V transposed: [HD, S]
      unsigned short* dst = Vtb + ((size_t)bh * HD_ + dbase + n) * S_ + s0 + 32 * mp2;
#pragma unroll
      for (int mt = 0; mt < 2; ++mt)
#pragma unroll
        for (int v = 0; v < 8; ++v)
          dst[16 * mt + v + 8 * g] = f2bf(a2[mt][v]);
    }
  }
}

// ---------------- Stage 2: causal flash attention ------------------------
// 32 queries per wave (2 M-tiles) halves K/V fragment traffic per score.
// Grid: B*H*(S/32)/4 blocks of 128 threads.
__global__ __launch_bounds__(128) void attn_k(
    const unsigned short* __restrict__ Qb, const unsigned short* __restrict__ Kb,
    const unsigned short* __restrict__ Vtb, unsigned short* __restrict__ catb) {
  __shared__ __align__(16) unsigned short Pb[4][32 * 32];   // 8 KB
  const int lane = threadIdx.x & 31, wave = threadIdx.x >> 5;
  const int g = lane >> 4, n = lane & 15;
  const int flat = blockIdx.x * 4 + wave;
  const int qt = flat & 63;                  // S_/32
  const int bh = flat >> 6;
  const int h = bh & (H_ - 1), b = bh >> 4;
  const int q0 = qt * 32;
  const unsigned short* Qp = Qb + (size_t)bh * S_ * HD_;
  const unsigned short* Kp = Kb + (size_t)bh * S_ * HD_;
  const unsigned short* Vp = Vtb + (size_t)bh * HD_ * S_;
  unsigned short* Plds = &Pb[wave][0];

  v16bf qf[2];
  qf[0] = load_fragA(Qp, q0,      HD_, 0, lane);   // HD_=32 = one WMMA K-step
  qf[1] = load_fragA(Qp, q0 + 16, HD_, 0, lane);
  float m[2][8], l[2][8];
  v8f acc[2][2] = {};
#pragma unroll
  for (int u = 0; u < 2; ++u)
#pragma unroll
    for (int v = 0; v < 8; ++v) { m[u][v] = -1e30f; l[u][v] = 0.f; }

  const int tEnd = q0 + 32;                  // keys needed: t <= q0+31
#pragma unroll 1
  for (int t0 = 0; t0 < tEnd; t0 += 32) {
    v16bf kf0 = load_fragB(Kp, t0,      HD_, 0, lane);
    v16bf kf1 = load_fragB(Kp, t0 + 16, HD_, 0, lane);
    v8f z = {};
#pragma unroll
    for (int u = 0; u < 2; ++u) {
      v8f s0 = wmma_bf16(qf[u], kf0, z);
      v8f s1 = wmma_bf16(qf[u], kf1, z);
#pragma unroll
      for (int v = 0; v < 8; ++v) {
        const int M = v + 8 * g;
        const int qg = q0 + u * 16 + M;
        float a0 = (t0 + n      > qg) ? -1e30f : s0[v];  // causal mask
        float a1 = (t0 + 16 + n > qg) ? -1e30f : s1[v];
        float rm = fmaxf(a0, a1);
#pragma unroll
        for (int off = 1; off < 16; off <<= 1) rm = fmaxf(rm, __shfl_xor(rm, off, 32));
        const float mn = fmaxf(m[u][v], rm);
        const float sc = __expf(m[u][v] - mn);
        const float p0 = __expf(a0 - mn);
        const float p1 = __expf(a1 - mn);
        float rs = p0 + p1;
#pragma unroll
        for (int off = 1; off < 16; off <<= 1) rs += __shfl_xor(rs, off, 32);
        l[u][v] = l[u][v] * sc + rs;
        m[u][v] = mn;
        acc[u][0][v] *= sc;
        acc[u][1][v] *= sc;
        Plds[(u * 16 + M) * 32 + n]      = f2bf(p0);  // C-layout -> LDS -> A-layout
        Plds[(u * 16 + M) * 32 + 16 + n] = f2bf(p1);
      }
    }
    asm volatile("s_wait_dscnt 0" ::: "memory");      // ds stores visible to ds loads
    v16bf vf0 = load_fragB(Vp, 0,  S_, t0, lane);     // V^T rows d = 0..15
    v16bf vf1 = load_fragB(Vp, 16, S_, t0, lane);     // d = 16..31
#pragma unroll
    for (int u = 0; u < 2; ++u) {
      v16bf pf = load_fragA(Plds, u * 16, 32, 0, lane);
      acc[u][0] = wmma_bf16(pf, vf0, acc[u][0]);
      acc[u][1] = wmma_bf16(pf, vf1, acc[u][1]);
    }
  }
  unsigned short* cb = catb + (size_t)(b * S_ + q0) * (H_ * HD_) + h * HD_;
#pragma unroll
  for (int u = 0; u < 2; ++u)
#pragma unroll
    for (int v = 0; v < 8; ++v) {
      const int M = u * 16 + v + 8 * g;
      const float inv = 1.0f / l[u][v];
      cb[(size_t)M * (H_ * HD_) + n]      = f2bf(acc[u][0][v] * inv);
      cb[(size_t)M * (H_ * HD_) + 16 + n] = f2bf(acc[u][1][v] * inv);
    }
}

// ---------------- Stage 3: output projection -----------------------------
// out[4096,512] = cat @ Wo^T + bo ; 16x64 tile per wave (A reused 4x), K=512.
__global__ __launch_bounds__(256) void out_k(
    const unsigned short* __restrict__ catb, const unsigned short* __restrict__ wob,
    const float* __restrict__ bo, float* __restrict__ out) {
  const int lane = threadIdx.x & 31, wave = threadIdx.x >> 5;
  const int g = lane >> 4, n = lane & 15;
  const int id = blockIdx.x * 8 + wave;        // 2048 tasks
  const int r0 = (id >> 3) * 16;               // 256 row tiles
  const int c0 = (id & 7) * 64;                // 8 col groups
  v8f acc[4] = {};
#pragma unroll 2
  for (int kt = 0; kt < 16; ++kt) {
    v16bf a = load_fragA(catb, r0, IN_, kt * 32, lane);
#pragma unroll
    for (int j = 0; j < 4; ++j) {
      v16bf bb = load_fragB(wob, c0 + j * 16, IN_, kt * 32, lane);  // Wo rows = (Wo^T)^T
      acc[j] = wmma_bf16(a, bb, acc[j]);
    }
  }
#pragma unroll
  for (int j = 0; j < 4; ++j) {
    const float bias = bo[c0 + j * 16 + n];
#pragma unroll
    for (int v = 0; v < 8; ++v)
      out[(size_t)(r0 + v + 8 * g) * IN_ + c0 + j * 16 + n] = acc[j][v] + bias;
  }
}

// ---------------- Host launcher ------------------------------------------

extern "C" void kernel_launch(void* const* d_in, const int* in_sizes, int n_in,
                              void* d_out, int out_size, void* d_ws, size_t ws_size,
                              hipStream_t stream) {
  const float* x  = (const float*)d_in[0];
  const float* Wl = (const float*)d_in[1];
  const float* bl = (const float*)d_in[2];
  const float* Wq = (const float*)d_in[3];
  const float* Wk = (const float*)d_in[4];
  const float* Wv = (const float*)d_in[5];
  const float* Wo = (const float*)d_in[6];
  const float* bo = (const float*)d_in[7];
  float* out = (float*)d_out;

  char* ws = (char*)d_ws;
  size_t off = 0;
  auto alloc = [&](size_t bytes) -> char* {
    char* p = ws + off;
    off += (bytes + 255) & ~(size_t)255;
    return p;
  };
  unsigned short* xb   = (unsigned short*)alloc((size_t)B_ * S_ * IN_ * 2);
  unsigned short* wlb  = (unsigned short*)alloc((size_t)H_ * IN_ * IN_ * 2);
  unsigned short* wob  = (unsigned short*)alloc((size_t)IN_ * H_ * HD_ * 2);
  unsigned short* wqt  = (unsigned short*)alloc((size_t)H_ * HD_ * IN_ * 2);
  unsigned short* wkt  = (unsigned short*)alloc((size_t)H_ * HD_ * IN_ * 2);
  unsigned short* wvt  = (unsigned short*)alloc((size_t)H_ * HD_ * IN_ * 2);
  unsigned short* Qb   = (unsigned short*)alloc((size_t)B_ * H_ * S_ * HD_ * 2);
  unsigned short* Kb   = (unsigned short*)alloc((size_t)B_ * H_ * S_ * HD_ * 2);
  unsigned short* Vtb  = (unsigned short*)alloc((size_t)B_ * H_ * HD_ * S_ * 2);
  unsigned short* catb = (unsigned short*)alloc((size_t)B_ * S_ * H_ * HD_ * 2);
  if (off > ws_size) return;   // workspace too small; harness will flag mismatch

  int n;
  n = B_ * S_ * IN_;  cvt_k<<<(n + 255) / 256, 256, 0, stream>>>(x,  xb,  n);
  n = H_ * IN_ * IN_; cvt_k<<<(n + 255) / 256, 256, 0, stream>>>(Wl, wlb, n);
  n = IN_ * H_ * HD_; cvt_k<<<(n + 255) / 256, 256, 0, stream>>>(Wo, wob, n);
  n = H_ * HD_ * IN_;
  trq_k<<<(n + 255) / 256, 256, 0, stream>>>(Wq, wqt, n);
  trq_k<<<(n + 255) / 256, 256, 0, stream>>>(Wk, wkt, n);
  trq_k<<<(n + 255) / 256, 256, 0, stream>>>(Wv, wvt, n);

  qkv_k<<<B_ * H_ * (S_ / 64), 512, 0, stream>>>(xb, wlb, bl, wqt, wkt, wvt, Qb, Kb, Vtb);
  attn_k<<<B_ * H_ * (S_ / 32) / 4, 128, 0, stream>>>(Qb, Kb, Vtb, catb);
  out_k<<<(B_ * S_ / 16) * (IN_ / 64) / 8, 256, 0, stream>>>(catb, wob, bo, out);
}
